// BaseConvolution_61976378081889
// MI455X (gfx1250) — matrix operand order repaired and checked by reading, not compiled, not run on hardware
//
#include <hip/hip_runtime.h>
#include <hip/hip_bf16.h>

typedef __attribute__((ext_vector_type(16))) _Float16 v16h;
typedef __attribute__((ext_vector_type(8)))  _Float16 v8h;
typedef __attribute__((ext_vector_type(8)))  float    v8f;
typedef __attribute__((ext_vector_type(4)))  int      v4i;

#define NB     8
#define NPER   4096
#define NSAMP  1024
#define KNBR   64
#define CIN    64
#define HID    128
#define KPAD   96        // 67 padded to 3 * 32 for WMMA k-steps
#define RAD2   0.04f

#if __has_builtin(__builtin_amdgcn_global_load_async_to_lds_b128) && \
    __has_builtin(__builtin_amdgcn_s_wait_asynccnt)
#define USE_ASYNC_LDS 1
#else
#define USE_ASYNC_LDS 0
#endif

#define AS_GLOBAL __attribute__((address_space(1)))
#define AS_LDS    __attribute__((address_space(3)))

static __device__ __forceinline__ v16h cat16(v8h lo, v8h hi) {
  return __builtin_shufflevector(lo, hi, 0, 1, 2, 3, 4, 5, 6, 7,
                                 8, 9, 10, 11, 12, 13, 14, 15);
}

// ---------------------------------------------------------------------------
// Kernel 0: W1 (67x128) -> f16 TRANSPOSED [128][96] (zero-pad K 67..95),
//           W2 (128x128) -> f16 TRANSPOSED [128][128]; batch_out tail.
// ---------------------------------------------------------------------------
__global__ void prep_kernel(const float* __restrict__ W1,
                            const float* __restrict__ W2,
                            _Float16* __restrict__ W1T,
                            _Float16* __restrict__ W2T,
                            float* __restrict__ d_out) {
  int i = blockIdx.x * 256 + threadIdx.x;
  if (i < HID * KPAD) {
    int n = i / KPAD, k = i % KPAD;
    W1T[i] = (_Float16)(k < CIN + 3 ? W1[k * HID + n] : 0.0f);
  }
  if (i < HID * HID) {
    int n = i / HID, k = i % HID;
    W2T[i] = (_Float16)W2[k * HID + n];
  }
  if (i < NB * NSAMP) {
    int* bout = (int*)(d_out + NB * NSAMP * HID + NB * NSAMP * 3);
    bout[i] = i >> 10;
  }
}

// ---------------------------------------------------------------------------
// Kernel 1: FPS. 1 block/cloud, 256 threads (8 waves). Serial-latency path:
// shfl_xor butterfly argmax per wave, 8 winners to LDS, 2 barriers per step.
// ---------------------------------------------------------------------------
__global__ void __launch_bounds__(256) fps_kernel(const float* __restrict__ pos,
                                                  int* __restrict__ idx_out,
                                                  float* __restrict__ pos_out) {
  __shared__ __align__(16) float px[NPER], py[NPER], pz[NPER];
  __shared__ float rv[8];
  __shared__ int   ri[8];
  const int b = blockIdx.x, t = threadIdx.x;
  const float* pb = pos + (size_t)b * NPER * 3;
  for (int j = t; j < NPER; j += 256) {
    px[j] = pb[j * 3 + 0]; py[j] = pb[j * 3 + 1]; pz[j] = pb[j * 3 + 2];
  }
  float md[16];
#pragma unroll
  for (int j = 0; j < 16; ++j) md[j] = 1e10f;
  __syncthreads();

  int last = 0;
  if (t == 0) {
    idx_out[b * NSAMP] = 0;
    pos_out[(size_t)(b * NSAMP) * 3 + 0] = px[0];
    pos_out[(size_t)(b * NSAMP) * 3 + 1] = py[0];
    pos_out[(size_t)(b * NSAMP) * 3 + 2] = pz[0];
  }
  const float4* px4 = (const float4*)px;
  const float4* py4 = (const float4*)py;
  const float4* pz4 = (const float4*)pz;
  const int wv = t >> 5, ln = t & 31;
  for (int s = 1; s < NSAMP; ++s) {
    const float lx = px[last], ly = py[last], lz = pz[last];
    float bv = -1.0f; int bi = 0;
#pragma unroll
    for (int q = 0; q < 4; ++q) {
      float4 X = px4[t * 4 + q], Y = py4[t * 4 + q], Z = pz4[t * 4 + q];
      float dx, dy, dz, d, m;
      int p0 = t * 16 + q * 4;
      dx = X.x - lx; dy = Y.x - ly; dz = Z.x - lz; d = dx*dx + dy*dy + dz*dz;
      m = fminf(md[q*4+0], d); md[q*4+0] = m; if (m > bv) { bv = m; bi = p0+0; }
      dx = X.y - lx; dy = Y.y - ly; dz = Z.y - lz; d = dx*dx + dy*dy + dz*dz;
      m = fminf(md[q*4+1], d); md[q*4+1] = m; if (m > bv) { bv = m; bi = p0+1; }
      dx = X.z - lx; dy = Y.z - ly; dz = Z.z - lz; d = dx*dx + dy*dy + dz*dz;
      m = fminf(md[q*4+2], d); md[q*4+2] = m; if (m > bv) { bv = m; bi = p0+2; }
      dx = X.w - lx; dy = Y.w - ly; dz = Z.w - lz; d = dx*dx + dy*dy + dz*dz;
      m = fminf(md[q*4+3], d); md[q*4+3] = m; if (m > bv) { bv = m; bi = p0+3; }
    }
#pragma unroll
    for (int off = 16; off > 0; off >>= 1) {
      float ov = __shfl_xor(bv, off, 32);
      int   oi = __shfl_xor(bi, off, 32);
      if (ov > bv || (ov == bv && oi < bi)) { bv = ov; bi = oi; }
    }
    if (ln == 0) { rv[wv] = bv; ri[wv] = bi; }
    __syncthreads();
    float fv = rv[0]; int fi = ri[0];
#pragma unroll
    for (int u = 1; u < 8; ++u) {
      float ov = rv[u]; int oi = ri[u];
      if (ov > fv || (ov == fv && oi < fi)) { fv = ov; fi = oi; }
    }
    last = fi;
    if (t == 0) {
      idx_out[b * NSAMP + s] = last;
      pos_out[(size_t)(b * NSAMP + s) * 3 + 0] = px[last];
      pos_out[(size_t)(b * NSAMP + s) * 3 + 1] = py[last];
      pos_out[(size_t)(b * NSAMP + s) * 3 + 2] = pz[last];
    }
    __syncthreads();
  }
}

// ---------------------------------------------------------------------------
// Kernel 2: top-64 query. block=64 (2 waves), wave-per-center.
// ---------------------------------------------------------------------------
__global__ void __launch_bounds__(64) ballq_kernel(const float* __restrict__ pos,
                                                   const int* __restrict__ idx,
                                                   int* __restrict__ nbr,
                                                   int* __restrict__ vld) {
  __shared__ __align__(16) float d2s[2][NPER];
  const int wave = threadIdx.x >> 5, lane = threadIdx.x & 31;
  const int c = blockIdx.x * 2 + wave;
  const int b = c >> 10;
  const float* pb = pos + (size_t)b * NPER * 3;
  const int ci = idx[c];
  const float cx = pb[ci * 3 + 0], cy = pb[ci * 3 + 1], cz = pb[ci * 3 + 2];
  float* d2 = d2s[wave];
  for (int j = 0; j < 128; ++j) {
    int p = lane * 128 + j;
    float dx = pb[p * 3 + 0] - cx, dy = pb[p * 3 + 1] - cy, dz = pb[p * 3 + 2] - cz;
    d2[p] = dx * dx + dy * dy + dz * dz;
  }
  __syncthreads();
  const float4* d24 = (const float4*)(d2 + lane * 128);
  for (int r = 0; r < KNBR; ++r) {
    float bv = 3.0e38f; int bi = 0x7fffffff;
    for (int q = 0; q < 32; ++q) {
      float4 v = d24[q];
      int p0 = lane * 128 + q * 4;
      if (v.x < bv) { bv = v.x; bi = p0 + 0; }
      if (v.y < bv) { bv = v.y; bi = p0 + 1; }
      if (v.z < bv) { bv = v.z; bi = p0 + 2; }
      if (v.w < bv) { bv = v.w; bi = p0 + 3; }
    }
#pragma unroll
    for (int off = 16; off > 0; off >>= 1) {
      float ov = __shfl_xor(bv, off, 32);
      int   oi = __shfl_xor(bi, off, 32);
      if (ov < bv || (ov == bv && oi < bi)) { bv = ov; bi = oi; }
    }
    if (lane == 0) {
      nbr[(size_t)c * KNBR + r] = bi;
      vld[(size_t)c * KNBR + r] = (bv <= RAD2) ? 1 : 0;
      d2[bi] = 3.0e38f;
    }
    __syncthreads();
  }
}

// ---------------------------------------------------------------------------
// Kernel 3: gather + 2-layer MLP (v_wmma_f32_16x16x32_f16) + masked max.
// Weight staging uses CDNA5 async global->LDS when available.
// ---------------------------------------------------------------------------
__global__ void __launch_bounds__(256) mlp_kernel(const float* __restrict__ x,
                                                  const float* __restrict__ pos,
                                                  const float* __restrict__ b1,
                                                  const float* __restrict__ b2,
                                                  const int* __restrict__ idx,
                                                  const int* __restrict__ nbr,
                                                  const int* __restrict__ vld,
                                                  const _Float16* __restrict__ W1T,
                                                  const _Float16* __restrict__ W2T,
                                                  float* __restrict__ xout) {
  __shared__ __align__(16) _Float16 sW[HID * HID];      // 32KB: W1T then W2T
  __shared__ __align__(16) _Float16 sFeat[KNBR * KPAD]; // 12KB row-major [64][96]
  __shared__ __align__(16) _Float16 sH1[KNBR * HID];    // 16KB row-major [64][128]
  __shared__ unsigned sMax[HID];
  __shared__ int   sNbr[KNBR], sVld[KNBR];
  __shared__ float sC[3];

  const int t = threadIdx.x;
  const int g = blockIdx.x;
  const int b = g >> 10;

  // stage W1T: 96*128 halves = 1536 x 16B
#if USE_ASYNC_LDS
  for (int i = t; i < (HID * KPAD * 2) / 16; i += 256) {
    __builtin_amdgcn_global_load_async_to_lds_b128(
        (AS_GLOBAL v4i*)((v4i*)W1T + i),
        (AS_LDS v4i*)((v4i*)sW + i), 0, 0);
  }
  __builtin_amdgcn_s_wait_asynccnt(0);
#else
  {
    const uint4* wvv = (const uint4*)W1T;
    uint4* sv = (uint4*)sW;
    for (int i = t; i < (HID * KPAD * 2) / 16; i += 256) sv[i] = wvv[i];
  }
#endif
  if (t < KNBR) { sNbr[t] = nbr[(size_t)g * KNBR + t]; sVld[t] = vld[(size_t)g * KNBR + t]; }
  if (t == 0) {
    int ci = idx[g];
    const float* pb = pos + (size_t)b * NPER * 3;
    sC[0] = pb[ci * 3 + 0]; sC[1] = pb[ci * 3 + 1]; sC[2] = pb[ci * 3 + 2];
  }
  if (t < HID) sMax[t] = 0u;
  __syncthreads();

  // build feature tile [64 x 96] in 8-half chunks (ds_store_b128)
  for (int ch = t; ch < KNBR * (KPAD / 8); ch += 256) {
    int row = ch / (KPAD / 8), c0 = (ch % (KPAD / 8)) * 8;
    int gi = b * NPER + sNbr[row];
    v8h o = (v8h)(_Float16)0.0f;
    if (c0 < CIN) {
      const float4* xp = (const float4*)&x[(size_t)gi * CIN + c0];
      float4 f0 = xp[0], f1 = xp[1];
      o[0] = (_Float16)f0.x; o[1] = (_Float16)f0.y;
      o[2] = (_Float16)f0.z; o[3] = (_Float16)f0.w;
      o[4] = (_Float16)f1.x; o[5] = (_Float16)f1.y;
      o[6] = (_Float16)f1.z; o[7] = (_Float16)f1.w;
    } else if (c0 == CIN) {
      o[0] = (_Float16)(pos[(size_t)gi * 3 + 0] - sC[0]);
      o[1] = (_Float16)(pos[(size_t)gi * 3 + 1] - sC[1]);
      o[2] = (_Float16)(pos[(size_t)gi * 3 + 2] - sC[2]);
    }
    *(v8h*)&sFeat[row * KPAD + c0] = o;
  }
  __syncthreads();

  const int w = t >> 5, lane = t & 31;
  const int row0 = (w >> 1) * 16;
  const int nbase = (w & 1) * 64;
  const int m = lane & 15, hi = lane >> 4;

  // ---- layer 1: [64x96] x [96x128] ----
  v16h a1[3];
  {
    const _Float16* ar = &sFeat[(row0 + m) * KPAD];
#pragma unroll
    for (int ks = 0; ks < 3; ++ks) {
      int k0 = ks * 32;
      v8h lo = *(const v8h*)&ar[k0 + hi * 8];
      v8h hh = *(const v8h*)&ar[k0 + 16 + hi * 8];
      a1[ks] = cat16(lo, hh);
    }
  }
#pragma unroll
  for (int nt = 0; nt < 4; ++nt) {
    int n0 = nbase + nt * 16;
    float bias = b1[n0 + m];
    v8f c;
#pragma unroll
    for (int r = 0; r < 8; ++r) c[r] = bias;
    const _Float16* wr = &sW[(n0 + m) * KPAD];
#pragma unroll
    for (int ks = 0; ks < 3; ++ks) {
      int k0 = ks * 32;
      v8h blo = *(const v8h*)&wr[k0 + hi * 8];
      v8h bhi = *(const v8h*)&wr[k0 + 16 + hi * 8];
      c = __builtin_amdgcn_wmma_f32_16x16x32_f16(false, a1[ks], false,
                                                 cat16(blo, bhi),
                                                 (short)0, c, false, false);
    }
#pragma unroll
    for (int r = 0; r < 8; ++r) {
      int row = row0 + r + hi * 8;
      sH1[row * HID + n0 + m] = (_Float16)fmaxf(c[r], 0.0f);
    }
  }
  __syncthreads();

  // stage W2T: 128*128 halves = 2048 x 16B (over the W1T region)
#if USE_ASYNC_LDS
  for (int i = t; i < (HID * HID * 2) / 16; i += 256) {
    __builtin_amdgcn_global_load_async_to_lds_b128(
        (AS_GLOBAL v4i*)((v4i*)W2T + i),
        (AS_LDS v4i*)((v4i*)sW + i), 0, 0);
  }
  __builtin_amdgcn_s_wait_asynccnt(0);
#else
  {
    const uint4* wvv = (const uint4*)W2T;
    uint4* sv = (uint4*)sW;
    for (int i = t; i < (HID * HID * 2) / 16; i += 256) sv[i] = wvv[i];
  }
#endif
  __syncthreads();

  // ---- layer 2: [64x128] x [128x128] + masked max over K rows ----
  v16h a2[4];
  {
    const _Float16* ar = &sH1[(row0 + m) * HID];
#pragma unroll
    for (int ks = 0; ks < 4; ++ks) {
      int k0 = ks * 32;
      v8h lo = *(const v8h*)&ar[k0 + hi * 8];
      v8h hh = *(const v8h*)&ar[k0 + 16 + hi * 8];
      a2[ks] = cat16(lo, hh);
    }
  }
#pragma unroll
  for (int nt = 0; nt < 4; ++nt) {
    int n0 = nbase + nt * 16;
    float bias = b2[n0 + m];
    v8f c;
#pragma unroll
    for (int r = 0; r < 8; ++r) c[r] = bias;
    const _Float16* wr = &sW[(n0 + m) * HID];
#pragma unroll
    for (int ks = 0; ks < 4; ++ks) {
      int k0 = ks * 32;
      v8h blo = *(const v8h*)&wr[k0 + hi * 8];
      v8h bhi = *(const v8h*)&wr[k0 + 16 + hi * 8];
      c = __builtin_amdgcn_wmma_f32_16x16x32_f16(false, a2[ks], false,
                                                 cat16(blo, bhi),
                                                 (short)0, c, false, false);
    }
#pragma unroll
    for (int r = 0; r < 8; ++r) {
      int row = row0 + r + hi * 8;
      float v = fmaxf(c[r], 0.0f);
      if (!sVld[row]) v = 0.0f;                  // == -1e9 mask (max >= 0 always)
      atomicMax(&sMax[n0 + m], __float_as_uint(v));
    }
  }
  __syncthreads();
  if (t < HID) xout[(size_t)g * HID + t] = __uint_as_float(sMax[t]);
}

// ---------------------------------------------------------------------------
extern "C" void kernel_launch(void* const* d_in, const int* in_sizes, int n_in,
                              void* d_out, int out_size, void* d_ws, size_t ws_size,
                              hipStream_t stream) {
  const float* x   = (const float*)d_in[0];
  const float* pos = (const float*)d_in[1];
  const float* W1  = (const float*)d_in[2];
  const float* b1  = (const float*)d_in[3];
  const float* W2  = (const float*)d_in[4];
  const float* b2  = (const float*)d_in[5];

  float* out = (float*)d_out;
  float* pos_out = out + (size_t)NB * NSAMP * HID;

  char* ws = (char*)d_ws;
  int*      idx = (int*)ws;                                   // 8192 ints
  int*      nbr = (int*)(ws + 32768);                         // 8192*64 ints
  int*      vld = (int*)(ws + 32768 + 2097152);               // 8192*64 ints
  _Float16* W1T = (_Float16*)(ws + 32768 + 2 * 2097152);      // 128*96 halves
  _Float16* W2T = (_Float16*)(ws + 32768 + 2 * 2097152 + 24576);

  prep_kernel<<<64, 256, 0, stream>>>(W1, W2, W1T, W2T, out);
  fps_kernel<<<NB, 256, 0, stream>>>(pos, idx, pos_out);
  ballq_kernel<<<(NB * NSAMP) / 2, 64, 0, stream>>>(pos, idx, nbr, vld);
  mlp_kernel<<<NB * NSAMP, 256, 0, stream>>>(x, pos, b1, b2, idx, nbr, vld,
                                             W1T, W2T, out);
}
